// LocalCrossAttention_25881472926453
// MI455X (gfx1250) — matrix operand order repaired and checked by reading, not compiled, not run on hardware
//
#include <hip/hip_runtime.h>
#include <hip/hip_bf16.h>

// Problem constants (match reference)
#define N1 40000
#define N2 60000
#define DD 128
#define HH 8
#define HD 16
#define KK 32
#define SCALE 0.25f   // HD^-0.5 = 16^-0.5

typedef __bf16 bf16_t;
typedef __attribute__((ext_vector_type(16))) __bf16 v16bf;
typedef __attribute__((ext_vector_type(8)))  __bf16 v8bf;
typedef __attribute__((ext_vector_type(8)))  float  v8f;

__device__ __forceinline__ unsigned short f32_to_bf16_rne(float f) {
    unsigned int u = __float_as_uint(f);
    unsigned int r = u + 0x7FFFu + ((u >> 16) & 1u);
    return (unsigned short)(r >> 16);
}

// ---------------------------------------------------------------------------
// Elementwise f32 -> bf16 (4 elements / thread, b128 in, b64 out)
// ---------------------------------------------------------------------------
__global__ void cvt_f32_bf16_x4(const float4* __restrict__ src,
                                ushort4* __restrict__ dst, int n4) {
    int i = blockIdx.x * blockDim.x + threadIdx.x;
    if (i < n4) {
        float4 f = src[i];
        ushort4 o;
        o.x = f32_to_bf16_rne(f.x);
        o.y = f32_to_bf16_rne(f.y);
        o.z = f32_to_bf16_rne(f.z);
        o.w = f32_to_bf16_rne(f.w);
        dst[i] = o;
    }
}

// ---------------------------------------------------------------------------
// 128x128 weight: f32 row-major W[k][n] -> bf16 transposed Wt[n][k]
// (so WMMA B-matrix column loads are contiguous)
// ---------------------------------------------------------------------------
__global__ void cvt_w_transpose(const float* __restrict__ W,
                                unsigned short* __restrict__ Wt) {
    int t = blockIdx.x * blockDim.x + threadIdx.x;   // 0..16383
    int n = t >> 7;
    int k = t & 127;
    Wt[t] = f32_to_bf16_rne(W[k * DD + n]);
}

// ---------------------------------------------------------------------------
// Y[M x 128] = Xb[M x 128](bf16) @ W(bf16, pre-transposed: Wt[n][k]) + bias
// One wave -> one 16x16 output tile. 8 waves/block cover N=128 for 16 rows.
// grid.x = M/16, block = 256 (8 waves). M must be a multiple of 16.
// A layout (16-bit 16x32, wave32): lane L (m=L&15, half=L>>4) holds
//   K = kb + half*16 + 0..15  -> two b128 loads.
// B layout (32x16): lane L holds column n=L&15, K = kb + half*16 + 0..15.
// C/D: lane L, vgpr r -> (m = r + 8*half, n = L&15).
// ---------------------------------------------------------------------------
__global__ void __launch_bounds__(256)
gemm128_bias_wmma(const unsigned short* __restrict__ Xb,
                  const unsigned short* __restrict__ Wtb,
                  const float* __restrict__ bias,
                  float* __restrict__ Y, int M) {
    const int lane  = threadIdx.x & 31;
    const int wv    = threadIdx.x >> 5;      // N-tile 0..7
    const int rowB  = blockIdx.x << 4;
    const int mr    = lane & 15;
    const int half  = lane >> 4;
    const int ncol0 = wv << 4;

    const bf16_t* X  = reinterpret_cast<const bf16_t*>(Xb);
    const bf16_t* Wt = reinterpret_cast<const bf16_t*>(Wtb);

    const bf16_t* arow = X  + (size_t)(rowB + mr) * DD;
    const bf16_t* bcol = Wt + (size_t)(ncol0 + mr) * DD + (half << 4);

    v8f acc = {};
#pragma unroll
    for (int kb = 0; kb < DD; kb += 32) {
        v8bf alo = *reinterpret_cast<const v8bf*>(arow + kb + (half << 4));
        v8bf ahi = *reinterpret_cast<const v8bf*>(arow + kb + 8 + (half << 4));
        v8bf blo = *reinterpret_cast<const v8bf*>(bcol + kb);
        v8bf bhi = *reinterpret_cast<const v8bf*>(bcol + kb + 8);
        v16bf a, b;
#pragma unroll
        for (int i = 0; i < 8; ++i) {
            a[i] = alo[i]; a[i + 8] = ahi[i];
            b[i] = blo[i]; b[i + 8] = bhi[i];
        }
        acc = __builtin_amdgcn_wmma_f32_16x16x32_bf16(
            false, a, false, b, (short)0, acc, false, false);
    }

    const float bn = bias[ncol0 + mr];
#pragma unroll
    for (int r = 0; r < 8; ++r) {
        int row = rowB + r + (half << 3);
        Y[(size_t)row * DD + ncol0 + mr] = acc[r] + bn;
    }
}

// ---------------------------------------------------------------------------
// kNN local attention: one thread per (query, head).
// Qp/Kp/Vp are f32 [N x 128]; output attended in bf16 [N1 x 128]
// (bf16 so it can feed the final WMMA output projection).
// ---------------------------------------------------------------------------
__global__ void __launch_bounds__(256)
knn_attention(const float* __restrict__ Qp,
              const float* __restrict__ Kp,
              const float* __restrict__ Vp,
              const long long* __restrict__ knn,
              unsigned short* __restrict__ attended) {
    int t = blockIdx.x * blockDim.x + threadIdx.x;
    if (t >= N1 * HH) return;
    const int nq = t >> 3;
    const int h  = t & 7;

    float q[HD];
    const float4* q4 = reinterpret_cast<const float4*>(Qp + (size_t)nq * DD + h * HD);
#pragma unroll
    for (int i = 0; i < 4; ++i) {
        float4 v = q4[i];
        q[4 * i + 0] = v.x; q[4 * i + 1] = v.y;
        q[4 * i + 2] = v.z; q[4 * i + 3] = v.w;
    }

    const long long* idxp = knn + (size_t)nq * KK;
    int   idxs[KK];
    float sc[KK];
    float mx = -3.0e38f;
#pragma unroll
    for (int k = 0; k < KK; ++k) {
        int idx = (int)idxp[k];
        idxs[k] = idx;
        const float4* kp4 = reinterpret_cast<const float4*>(Kp + (size_t)idx * DD + h * HD);
        float s = 0.0f;
#pragma unroll
        for (int i = 0; i < 4; ++i) {
            float4 v = kp4[i];
            s += q[4 * i + 0] * v.x + q[4 * i + 1] * v.y
               + q[4 * i + 2] * v.z + q[4 * i + 3] * v.w;
        }
        s *= SCALE;
        sc[k] = s;
        mx = fmaxf(mx, s);
    }

    float denom = 0.0f;
#pragma unroll
    for (int k = 0; k < KK; ++k) {
        float e = __expf(sc[k] - mx);
        sc[k] = e;
        denom += e;
    }
    const float inv = 1.0f / denom;

    float acc[HD];
#pragma unroll
    for (int d = 0; d < HD; ++d) acc[d] = 0.0f;
#pragma unroll
    for (int k = 0; k < KK; ++k) {
        const float w = sc[k] * inv;
        const float4* vp4 = reinterpret_cast<const float4*>(Vp + (size_t)idxs[k] * DD + h * HD);
#pragma unroll
        for (int i = 0; i < 4; ++i) {
            float4 v = vp4[i];
            acc[4 * i + 0] += w * v.x; acc[4 * i + 1] += w * v.y;
            acc[4 * i + 2] += w * v.z; acc[4 * i + 3] += w * v.w;
        }
    }

    ushort4* o4 = reinterpret_cast<ushort4*>(attended + (size_t)nq * DD + h * HD);
#pragma unroll
    for (int i = 0; i < 4; ++i) {
        ushort4 o;
        o.x = f32_to_bf16_rne(acc[4 * i + 0]);
        o.y = f32_to_bf16_rne(acc[4 * i + 1]);
        o.z = f32_to_bf16_rne(acc[4 * i + 2]);
        o.w = f32_to_bf16_rne(acc[4 * i + 3]);
        o4[i] = o;
    }
}

// ---------------------------------------------------------------------------
extern "C" void kernel_launch(void* const* d_in, const int* in_sizes, int n_in,
                              void* d_out, int out_size, void* d_ws, size_t ws_size,
                              hipStream_t stream) {
    (void)in_sizes; (void)n_in; (void)out_size; (void)ws_size;

    const float*     Xq  = (const float*)d_in[0];
    const float*     Xk  = (const float*)d_in[1];
    const long long* knn = (const long long*)d_in[2];
    const float* Wq = (const float*)d_in[3];  const float* bq = (const float*)d_in[4];
    const float* Wk = (const float*)d_in[5];  const float* bk = (const float*)d_in[6];
    const float* Wv = (const float*)d_in[7];  const float* bv = (const float*)d_in[8];
    const float* Wo = (const float*)d_in[9];  const float* bo = (const float*)d_in[10];
    float* out = (float*)d_out;

    // Workspace carving (256-byte aligned slices)
    char* ws = (char*)d_ws;
    size_t off = 0;
    auto carve = [&](size_t bytes) -> char* {
        char* p = ws + off;
        off = (off + bytes + 255) & ~(size_t)255;
        return p;
    };
    unsigned short* Xq_b  = (unsigned short*)carve((size_t)N1 * DD * 2);
    unsigned short* Xk_b  = (unsigned short*)carve((size_t)N2 * DD * 2);
    unsigned short* Wq_t  = (unsigned short*)carve((size_t)DD * DD * 2);
    unsigned short* Wk_t  = (unsigned short*)carve((size_t)DD * DD * 2);
    unsigned short* Wv_t  = (unsigned short*)carve((size_t)DD * DD * 2);
    unsigned short* Wo_t  = (unsigned short*)carve((size_t)DD * DD * 2);
    float*          Qp    = (float*)carve((size_t)N1 * DD * 4);
    float*          Kp    = (float*)carve((size_t)N2 * DD * 4);
    float*          Vp    = (float*)carve((size_t)N2 * DD * 4);
    unsigned short* att_b = (unsigned short*)carve((size_t)N1 * DD * 2);

    // 1) Convert activations to bf16
    {
        int n4q = N1 * DD / 4;
        cvt_f32_bf16_x4<<<(n4q + 255) / 256, 256, 0, stream>>>(
            (const float4*)Xq, (ushort4*)Xq_b, n4q);
        int n4k = N2 * DD / 4;
        cvt_f32_bf16_x4<<<(n4k + 255) / 256, 256, 0, stream>>>(
            (const float4*)Xk, (ushort4*)Xk_b, n4k);
    }
    // 2) Convert + transpose weights to bf16
    cvt_w_transpose<<<64, 256, 0, stream>>>(Wq, Wq_t);
    cvt_w_transpose<<<64, 256, 0, stream>>>(Wk, Wk_t);
    cvt_w_transpose<<<64, 256, 0, stream>>>(Wv, Wv_t);
    cvt_w_transpose<<<64, 256, 0, stream>>>(Wo, Wo_t);

    // 3) WMMA projections: Q = Xq@Wq+bq, K = Xk@Wk+bk, V = Xk@Wv+bv
    gemm128_bias_wmma<<<N1 / 16, 256, 0, stream>>>(Xq_b, Wq_t, bq, Qp, N1);
    gemm128_bias_wmma<<<N2 / 16, 256, 0, stream>>>(Xk_b, Wk_t, bk, Kp, N2);
    gemm128_bias_wmma<<<N2 / 16, 256, 0, stream>>>(Xk_b, Wv_t, bv, Vp, N2);

    // 4) kNN local attention (L2-resident gathers), bf16 output
    knn_attention<<<(N1 * HH + 255) / 256, 256, 0, stream>>>(Qp, Kp, Vp, knn, att_b);

    // 5) WMMA output projection: out = attended @ Wo + bo (f32 result)
    gemm128_bias_wmma<<<N1 / 16, 256, 0, stream>>>(att_b, Wo_t, bo, out, N1);
}
